// BatchTopKCrosscoder_91173565760153
// MI455X (gfx1250) — matrix operand order repaired and checked by reading, not compiled, not run on hardware
//
#include <hip/hip_runtime.h>
#include <cstdint>

// ---------------------------------------------------------------------------
// BatchTopK crosscoder: encode (WMMA bf16x3) -> radix-select top k*B -> mask
// -> sparse decode.  B=2048, Z=1536, F=32768, k read from device.
// d_out layout (floats): recon[2048*1536] | sparse[2048*32768] | act[2048*32768]
// ---------------------------------------------------------------------------

#define Bdim 2048
#define Zdim 1536
#define Fdim 32768
#define NTOT (67108864LL)   // B*F

typedef __attribute__((ext_vector_type(8)))  float  v8f;
typedef __attribute__((ext_vector_type(16))) __bf16 v16bf;

union Frag { unsigned u[8]; v16bf v; };

struct SelState {
    unsigned prefix;   // accumulated high bits of threshold
    unsigned k_rem;    // remaining count to select inside current bin
    unsigned T;        // final threshold (float bits)
    unsigned E;        // how many ==T elements to accept
    unsigned C;        // total count of ==T elements
    unsigned strict;   // 1 if E==C (deterministic >= masking), else atomic ties
    unsigned tie;      // runtime tie counter
};

// float -> bf16 bits, round-to-nearest-even (finite inputs)
__device__ __forceinline__ unsigned short bf_bits(float f) {
    unsigned u = __float_as_uint(f);
    unsigned r = (u + 0x7FFFu + ((u >> 16) & 1u)) >> 16;
    return (unsigned short)r;
}
__device__ __forceinline__ float bf_to_f(unsigned short b) {
    return __uint_as_float(((unsigned)b) << 16);
}

// ---------------------------------------------------------------------------
// 1) decoder row norms: ||W_dec[f,:]||_2   (one block per feature)
// ---------------------------------------------------------------------------
__global__ __launch_bounds__(256) void norms_kernel(const float* __restrict__ Wdec,
                                                    float* __restrict__ norms) {
    const int f = blockIdx.x;
    const float* row = Wdec + (long long)f * Zdim;
    float s = 0.f;
    for (int i = threadIdx.x; i < Zdim; i += 256) { float w = row[i]; s += w * w; }
    __shared__ float red[256];
    red[threadIdx.x] = s;
    __syncthreads();
    for (int off = 128; off > 0; off >>= 1) {
        if (threadIdx.x < off) red[threadIdx.x] += red[threadIdx.x + off];
        __syncthreads();
    }
    if (threadIdx.x == 0) norms[f] = sqrtf(red[0]);
}

// ---------------------------------------------------------------------------
// 2) encode GEMM: act = relu(x @ W_enc + b), bf16x3 split precision WMMA.
//    Block tile 128(M) x 128(N), BK=32.  8 waves: 4 along M x 2 along N.
//    Each wave owns a 32x64 strip = 2x4 accumulators of 16x16 ->
//    24 WMMA per K-step with B-fragments reused across 2 M-tiles.
//    Next K-tile is prefetched into registers while current tile computes.
// ---------------------------------------------------------------------------
#define BM 128
#define BN 128
#define BK 32
#define LDK 34   // padded k-stride (shorts): 68B row stride, conflict-free

__global__ __launch_bounds__(256) void encode_kernel(const float* __restrict__ x,
                                                     const float* __restrict__ W,
                                                     const float* __restrict__ bias,
                                                     float* __restrict__ act) {
    __shared__ unsigned short As_hi[BM][LDK];
    __shared__ unsigned short As_lo[BM][LDK];
    __shared__ unsigned short Bs_hi[BN][LDK];   // stored transposed: [n][k]
    __shared__ unsigned short Bs_lo[BN][LDK];

    const int tid   = threadIdx.x;
    const int lane  = tid & 31;
    const int wave  = tid >> 5;
    const int wm    = wave & 3;        // 0..3  -> M offset (32 each)
    const int wn    = wave >> 2;       // 0..1  -> N offset (64 each)
    const int bm    = blockIdx.y * BM;
    const int bn    = blockIdx.x * BN;
    const int h     = lane >> 4;       // half-wave select
    const int l15   = lane & 15;

    v8f acc[2][4];
#pragma unroll
    for (int mt = 0; mt < 2; ++mt)
#pragma unroll
        for (int nt = 0; nt < 4; ++nt) acc[mt][nt] = v8f(0.f);

    // staging registers for the next K-tile (A: 128x32, B: 32x128; 16 ea/thread)
    float aReg[16], bReg[16];

    // ---- load first K-tile into registers ----
#pragma unroll
    for (int i = 0; i < 16; ++i) {
        int idx = i * 256 + tid;
        aReg[i] = x[(long long)(bm + (idx >> 5)) * Zdim + (idx & 31)];
        bReg[i] = W[(long long)(idx >> 7) * Fdim + bn + (idx & 127)];
    }
    // ---- convert + store to LDS ----
#pragma unroll
    for (int i = 0; i < 16; ++i) {
        int idx = i * 256 + tid;
        {
            int r = idx >> 5, c = idx & 31;
            unsigned short hi = bf_bits(aReg[i]);
            As_hi[r][c] = hi;
            As_lo[r][c] = bf_bits(aReg[i] - bf_to_f(hi));
        }
        {
            int r = idx >> 7, c = idx & 127;
            unsigned short hi = bf_bits(bReg[i]);
            Bs_hi[c][r] = hi;
            Bs_lo[c][r] = bf_bits(bReg[i] - bf_to_f(hi));
        }
    }
    __syncthreads();

    for (int kt = 0; kt < Zdim; kt += BK) {
        const bool more = (kt + BK) < Zdim;
        // ---- prefetch next K-tile into registers (hidden behind WMMA) ----
        if (more) {
#pragma unroll
            for (int i = 0; i < 16; ++i) {
                int idx = i * 256 + tid;
                aReg[i] = x[(long long)(bm + (idx >> 5)) * Zdim + kt + BK + (idx & 31)];
                bReg[i] = W[(long long)(kt + BK + (idx >> 7)) * Fdim + bn + (idx & 127)];
            }
        }

        // ---- compute on current LDS tile ----
        // A fragment per ISA layout: lanes 0-15 hold K[0:8)+K[16:24),
        // lanes 16-31 hold K[8:16)+K[24:32), row M = lane&15
        Frag a_hi[2], a_lo[2];
#pragma unroll
        for (int mt = 0; mt < 2; ++mt) {
            const int am = wm * 32 + mt * 16 + l15;
#pragma unroll
            for (int p = 0; p < 8; ++p) {
                int kk = (p < 4 ? 2 * p : 2 * p + 8) + h * 8;
                a_hi[mt].u[p] = *(const unsigned*)&As_hi[am][kk];
                a_lo[mt].u[p] = *(const unsigned*)&As_lo[am][kk];
            }
        }
#pragma unroll
        for (int nt = 0; nt < 4; ++nt) {
            // B fragment: lanes 0-15 K[0:8)+K[16:24), lanes 16-31 K[8:16)+K[24:32), col N=lane&15
            Frag b_hi, b_lo;
            const int nn = wn * 64 + nt * 16 + l15;
#pragma unroll
            for (int p = 0; p < 8; ++p) {
                int kk = (p < 4 ? 2 * p : 2 * p + 8) + h * 8;
                b_hi.u[p] = *(const unsigned*)&Bs_hi[nn][kk];
                b_lo.u[p] = *(const unsigned*)&Bs_lo[nn][kk];
            }
#pragma unroll
            for (int mt = 0; mt < 2; ++mt) {
                acc[mt][nt] = __builtin_amdgcn_wmma_f32_16x16x32_bf16(
                    false, a_hi[mt].v, false, b_hi.v, (short)0, acc[mt][nt], false, false);
                acc[mt][nt] = __builtin_amdgcn_wmma_f32_16x16x32_bf16(
                    false, a_hi[mt].v, false, b_lo.v, (short)0, acc[mt][nt], false, false);
                acc[mt][nt] = __builtin_amdgcn_wmma_f32_16x16x32_bf16(
                    false, a_lo[mt].v, false, b_hi.v, (short)0, acc[mt][nt], false, false);
            }
        }
        __syncthreads();

        // ---- convert + store prefetched tile to LDS ----
        if (more) {
#pragma unroll
            for (int i = 0; i < 16; ++i) {
                int idx = i * 256 + tid;
                {
                    int r = idx >> 5, c = idx & 31;
                    unsigned short hi = bf_bits(aReg[i]);
                    As_hi[r][c] = hi;
                    As_lo[r][c] = bf_bits(aReg[i] - bf_to_f(hi));
                }
                {
                    int r = idx >> 7, c = idx & 127;
                    unsigned short hi = bf_bits(bReg[i]);
                    Bs_hi[c][r] = hi;
                    Bs_lo[c][r] = bf_bits(bReg[i] - bf_to_f(hi));
                }
            }
            __syncthreads();
        }
    }

    // epilogue: bias + relu.  C/D layout: VGPR r -> M = r (+8 for lanes 16-31)
#pragma unroll
    for (int nt = 0; nt < 4; ++nt) {
        const int n_g = bn + wn * 64 + nt * 16 + l15;
        const float bv = bias[n_g];
#pragma unroll
        for (int mt = 0; mt < 2; ++mt) {
#pragma unroll
            for (int r = 0; r < 8; ++r) {
                const int m_g = bm + wm * 32 + mt * 16 + r + 8 * h;
                float v = acc[mt][nt][r] + bv;
                act[(long long)m_g * Fdim + n_g] = fmaxf(v, 0.f);
            }
        }
    }
}

// ---------------------------------------------------------------------------
// 3) radix select (3 passes: 12 + 12 + 8 bits, scores >= 0 so uint order works)
// ---------------------------------------------------------------------------
__global__ void init_state(SelState* st, const int* kptr) {
    if (blockIdx.x == 0 && threadIdx.x == 0) {
        st->prefix = 0u;
        st->k_rem  = (unsigned)(kptr[0] * Bdim);
        st->T = 0u; st->E = 0u; st->C = 0u; st->strict = 1u; st->tie = 0u;
    }
}

__global__ void zero_hist(unsigned* hist) {
    int i = blockIdx.x * 256 + threadIdx.x;
    if (i < 4096) hist[i] = 0u;
}

__global__ __launch_bounds__(256) void hist_kernel(const float* __restrict__ act,
                                                   const float* __restrict__ norms,
                                                   unsigned* __restrict__ hist,
                                                   const SelState* __restrict__ st,
                                                   int pass) {
    __shared__ unsigned lh[4096];
    for (int i = threadIdx.x; i < 4096; i += 256) lh[i] = 0u;
    __syncthreads();
    const unsigned prefix = st->prefix;
    const long long stride = (long long)gridDim.x * blockDim.x;
    for (long long i = (long long)blockIdx.x * blockDim.x + threadIdx.x;
         i < NTOT; i += stride) {
        const int f = (int)(i & (Fdim - 1));
        const unsigned u = __float_as_uint(act[i] * norms[f]);
        if (pass == 0) {
            atomicAdd(&lh[u >> 20], 1u);
        } else if (pass == 1) {
            if ((u >> 20) == (prefix >> 20)) atomicAdd(&lh[(u >> 8) & 0xFFFu], 1u);
        } else {
            if ((u >> 8) == (prefix >> 8)) atomicAdd(&lh[u & 0xFFu], 1u);
        }
    }
    __syncthreads();
    const int nb = (pass == 2) ? 256 : 4096;
    for (int i = threadIdx.x; i < nb; i += 256)
        if (lh[i]) atomicAdd(&hist[i], lh[i]);
}

__global__ void scan_kernel(const unsigned* __restrict__ hist, SelState* st,
                            int pass, int nbins, int shift) {
    if (blockIdx.x != 0 || threadIdx.x != 0) return;
    const unsigned krem = st->k_rem;
    unsigned long long cum = 0;
    int pick = 0;
    for (int b = nbins - 1; b >= 0; --b) {
        const unsigned c = hist[b];
        if (cum + c >= (unsigned long long)krem) { pick = b; break; }
        cum += c;
        if (b == 0) { pick = 0; cum -= c; }   // safety fallback
    }
    st->prefix |= ((unsigned)pick) << shift;
    st->k_rem = krem - (unsigned)cum;
    if (pass == 2) {
        st->T = st->prefix;
        st->E = st->k_rem;
        st->C = hist[pick];
        st->strict = (st->E >= st->C) ? 1u : 0u;
    }
}

// ---------------------------------------------------------------------------
// 4) mask: sparse = act where score in top k*B (exact count via tie budget E)
// ---------------------------------------------------------------------------
__global__ __launch_bounds__(256) void mask_kernel(const float* __restrict__ act,
                                                   const float* __restrict__ norms,
                                                   float* __restrict__ sparse,
                                                   SelState* st) {
    const unsigned T = st->T, E = st->E, strict = st->strict;
    const long long stride = (long long)gridDim.x * blockDim.x;
    for (long long i = (long long)blockIdx.x * blockDim.x + threadIdx.x;
         i < NTOT; i += stride) {
        const float a = act[i];
        const unsigned u = __float_as_uint(a * norms[(int)(i & (Fdim - 1))]);
        float out = 0.f;
        if (u > T) {
            out = a;
        } else if (u == T) {
            if (strict) out = a;
            else if (atomicAdd(&st->tie, 1u) < E) out = a;
        }
        sparse[i] = out;
    }
}

// ---------------------------------------------------------------------------
// 5) sparse decode: recon[b,:] = sum_f sparse[b,f] * W_dec[f,:] + b_dec.
//    One block per batch row; waves ballot-scan the sparse row and skip zeros.
// ---------------------------------------------------------------------------
__global__ __launch_bounds__(256) void decode_kernel(const float* __restrict__ sparse,
                                                     const float* __restrict__ Wdec,
                                                     const float* __restrict__ bdec,
                                                     float* __restrict__ recon) {
    __shared__ float red[8 * Zdim];
    const int b    = blockIdx.x;
    const int lane = threadIdx.x & 31;
    const int wave = threadIdx.x >> 5;

    float acc[48];
#pragma unroll
    for (int u = 0; u < 48; ++u) acc[u] = 0.f;

    const float* srow = sparse + (long long)b * Fdim;
    for (int base = wave * 32; base < Fdim; base += 256) {
        const float val = srow[base + lane];
        unsigned long long m = __ballot(val != 0.f);
        while (m) {
            const int j = __ffsll(m) - 1;
            m &= (m - 1);
            const float v = __shfl(val, j, 32);
            const float* wrow = Wdec + (long long)(base + j) * Zdim;
#pragma unroll
            for (int u = 0; u < 48; ++u) acc[u] += v * wrow[lane + 32 * u];
        }
    }
#pragma unroll
    for (int u = 0; u < 48; ++u) red[wave * Zdim + lane + 32 * u] = acc[u];
    __syncthreads();
    for (int c = threadIdx.x; c < Zdim; c += 256) {
        float s = 0.f;
#pragma unroll
        for (int w = 0; w < 8; ++w) s += red[w * Zdim + c];
        recon[(long long)b * Zdim + c] = s + bdec[c];
    }
}

// ---------------------------------------------------------------------------
extern "C" void kernel_launch(void* const* d_in, const int* in_sizes, int n_in,
                              void* d_out, int out_size, void* d_ws, size_t ws_size,
                              hipStream_t stream) {
    (void)in_sizes; (void)n_in; (void)out_size; (void)ws_size;

    const float* x    = (const float*)d_in[0];
    const float* Wenc = (const float*)d_in[1];
    const float* benc = (const float*)d_in[2];
    const float* Wdec = (const float*)d_in[3];
    const float* bdec = (const float*)d_in[4];
    const int*   kptr = (const int*)d_in[5];

    float* out    = (float*)d_out;
    float* recon  = out;                              // [2048*1536]
    float* sparse = out + (long long)Bdim * Zdim;     // [2048*32768]
    float* act    = sparse + NTOT;                    // [2048*32768]

    char*     ws    = (char*)d_ws;
    float*    norms = (float*)ws;                     // 32768 f32  (128 KB)
    unsigned* hist  = (unsigned*)(ws + 131072);       // 4096 u32   (16 KB)
    SelState* st    = (SelState*)(ws + 147456);

    // 1) decoder norms
    norms_kernel<<<Fdim, 256, 0, stream>>>(Wdec, norms);

    // 2) encode GEMM (WMMA bf16x3) -> activations
    dim3 eg(Fdim / BN, Bdim / BM);                    // 256 x 16 blocks
    encode_kernel<<<eg, 256, 0, stream>>>(x, Wenc, benc, act);

    // 3) radix select threshold for global top k*B
    init_state<<<1, 32, 0, stream>>>(st, kptr);
    const int nbins[3] = {4096, 4096, 256};
    const int shift[3] = {20, 8, 0};
    for (int pass = 0; pass < 3; ++pass) {
        zero_hist<<<16, 256, 0, stream>>>(hist);
        hist_kernel<<<2048, 256, 0, stream>>>(act, norms, hist, st, pass);
        scan_kernel<<<1, 32, 0, stream>>>(hist, st, pass, nbins[pass], shift[pass]);
    }

    // 4) mask -> sparse activations
    mask_kernel<<<8192, 256, 0, stream>>>(act, norms, sparse, st);

    // 5) sparse decode -> recon
    decode_kernel<<<Bdim, 256, 0, stream>>>(sparse, Wdec, bdec, recon);
}
